// MoE_75393855914557
// MI455X (gfx1250) — compile-verified, matrix operand order checked by reading, not run on hardware
//
#include <hip/hip_runtime.h>
#include <stdint.h>

typedef __attribute__((ext_vector_type(16))) __bf16 v16bf;
typedef __attribute__((ext_vector_type(8)))  float  v8f;

#define D_DIM 1024
#define T_DIM 16384
#define BM 128
#define BN 128
#define BK 32
#define NCHUNK (D_DIM / BK)

// ---- fp32 -> bf16 hi/lo split (RNE via native __bf16 casts) ----
__device__ __forceinline__ void split2(float f0, float f1, uint32_t& hi, uint32_t& lo) {
    __bf16 h0 = (__bf16)f0;
    __bf16 h1 = (__bf16)f1;
    float  r0 = f0 - (float)h0;
    float  r1 = f1 - (float)h1;
    __bf16 l0 = (__bf16)r0;
    __bf16 l1 = (__bf16)r1;
    union { __bf16 b[2]; uint32_t u; } H, L;
    H.b[0] = h0; H.b[1] = h1;
    L.b[0] = l0; L.b[1] = l1;
    hi = H.u; lo = L.u;
}

// ---- gfx1250 async global->LDS copy (ASYNCcnt-tracked, bypasses VGPRs) ----
// VDST VGPR holds the wave-relative LDS byte offset (= low 32 bits of the
// generic shared-aperture address); VADDR is the 64-bit global address.
__device__ __forceinline__ void async_ld_b128(void* lds_ptr, const void* gptr) {
    uint32_t lds_off = (uint32_t)(uintptr_t)lds_ptr;
    asm volatile("global_load_async_to_lds_b128 %0, %1, off"
                 :: "v"(lds_off), "v"(gptr)
                 : "memory");
}
__device__ __forceinline__ void wait_async0() {
    asm volatile("s_wait_asynccnt 0x0" ::: "memory");
}

// Load one 16x32 bf16 fragment from an LDS plane (row-major, 16 u32 per row).
// ISA 7.12.2: lanes 0-15 -> M/N = lane, K runs [0..7],[16..23];
//             lanes 16-31 -> M/N = lane-16, K runs [8..15],[24..31].
__device__ __forceinline__ v16bf load_frag(const uint32_t* plane, int rowBase, int lane) {
    int r  = rowBase + (lane & 15);
    int kB = (lane < 16) ? 0 : 4;            // u32 units
    const uint32_t* p = plane + r * 16 + kB;
    union { uint4 q[2]; v16bf v; } u;
    u.q[0] = *(const uint4*)(p);
    u.q[1] = *(const uint4*)(p + 8);
    return u.v;
}

// ---- prepass: Wc = W_shared + W_expert, split into bf16 hi/lo planes in ws ----
__global__ __launch_bounds__(256)
void prep_w_kernel(const float* __restrict__ Ws, const float* __restrict__ We,
                   uint32_t* __restrict__ wh, uint32_t* __restrict__ wl) {
    int i = blockIdx.x * 256 + threadIdx.x;          // one packed u32 (2 elements)
    const float2 a = ((const float2*)Ws)[i];
    const float2 b = ((const float2*)We)[i];
    uint32_t h, l;
    split2(a.x + b.x, a.y + b.y, h, l);
    wh[i] = h;
    wl[i] = l;
}

template<bool USE_WS>
__global__ __launch_bounds__(256)
void moe_gemm_kernel(const float* __restrict__ x,
                     const float* __restrict__ Ws,
                     const float* __restrict__ We,
                     const uint32_t* __restrict__ wh,
                     const uint32_t* __restrict__ wl,
                     const float* __restrict__ bias,
                     float* __restrict__ out) {
    // double-buffered: [buf][plane(Ahi,Alo,Bhi,Blo)][128 rows x 16 u32]
    __shared__ uint32_t lds[2][4][2048];             // 64 KB

    const int tid  = threadIdx.x;
    const int lane = tid & 31;
    const int wave = tid >> 5;
    const int wm   = wave & 3;                       // 4 waves x 32 rows (M)
    const int wn   = wave >> 2;                      // 2 waves x 64 cols (N)

    const int rowM0 = blockIdx.y * BM;
    const int rowN0 = blockIdx.x * BN;

    // A staging: one row (0..127) + one 16-element K half per thread
    const int srow = tid >> 1;
    const int skh  = (tid & 1) << 4;

    float4 xreg[4];
    float4 wsr[4], wer[4];                           // fallback path only

    auto loadA = [&](int kc) {
        const float4* xq = (const float4*)(x + (size_t)(rowM0 + srow) * D_DIM + kc + skh);
        xreg[0] = xq[0]; xreg[1] = xq[1]; xreg[2] = xq[2]; xreg[3] = xq[3];
    };
    auto storeA = [&](int buf) {
        uint32_t hi[8], lo[8];
        #pragma unroll
        for (int q = 0; q < 4; ++q) {
            split2(xreg[q].x, xreg[q].y, hi[2 * q + 0], lo[2 * q + 0]);
            split2(xreg[q].z, xreg[q].w, hi[2 * q + 1], lo[2 * q + 1]);
        }
        uint32_t* dh = &lds[buf][0][0] + srow * 16 + (skh >> 1);
        uint32_t* dl = &lds[buf][1][0] + srow * 16 + (skh >> 1);
        ((uint4*)dh)[0] = make_uint4(hi[0], hi[1], hi[2], hi[3]);
        ((uint4*)dh)[1] = make_uint4(hi[4], hi[5], hi[6], hi[7]);
        ((uint4*)dl)[0] = make_uint4(lo[0], lo[1], lo[2], lo[3]);
        ((uint4*)dl)[1] = make_uint4(lo[4], lo[5], lo[6], lo[7]);
    };
    // B staging, async path: pure bf16 copies from the pre-split planes.
    // 512 16-byte chunks per plane; thread covers chunks {tid, tid+256}.
    auto stageB_async = [&](int buf, int kc) {
        #pragma unroll
        for (int j = 0; j < 2; ++j) {
            int c    = tid + j * 256;
            int row  = c >> 2;
            int part = c & 3;
            size_t   src = ((size_t)(rowN0 + row) * D_DIM + kc) * 2 + part * 16; // bytes
            uint32_t dst = row * 64 + part * 16;                                  // bytes
            async_ld_b128((char*)&lds[buf][2][0] + dst, (const char*)wh + src);
            async_ld_b128((char*)&lds[buf][3][0] + dst, (const char*)wl + src);
        }
    };
    // B staging, fallback path: read fp32 W_shared/W_expert, add + split on the fly.
    auto loadB = [&](int kc) {
        const size_t woff = (size_t)(rowN0 + srow) * D_DIM + kc + skh;
        const float4* q0 = (const float4*)(Ws + woff);
        const float4* q1 = (const float4*)(We + woff);
        #pragma unroll
        for (int q = 0; q < 4; ++q) { wsr[q] = q0[q]; wer[q] = q1[q]; }
    };
    auto storeB = [&](int buf) {
        uint32_t hi[8], lo[8];
        #pragma unroll
        for (int q = 0; q < 4; ++q) {
            split2(wsr[q].x + wer[q].x, wsr[q].y + wer[q].y, hi[2 * q + 0], lo[2 * q + 0]);
            split2(wsr[q].z + wer[q].z, wsr[q].w + wer[q].w, hi[2 * q + 1], lo[2 * q + 1]);
        }
        uint32_t* dh = &lds[buf][2][0] + srow * 16 + (skh >> 1);
        uint32_t* dl = &lds[buf][3][0] + srow * 16 + (skh >> 1);
        ((uint4*)dh)[0] = make_uint4(hi[0], hi[1], hi[2], hi[3]);
        ((uint4*)dh)[1] = make_uint4(hi[4], hi[5], hi[6], hi[7]);
        ((uint4*)dl)[0] = make_uint4(lo[0], lo[1], lo[2], lo[3]);
        ((uint4*)dl)[1] = make_uint4(lo[4], lo[5], lo[6], lo[7]);
    };

    v8f acc[2][4];
    #pragma unroll
    for (int mi = 0; mi < 2; ++mi)
        #pragma unroll
        for (int ni = 0; ni < 4; ++ni)
            acc[mi][ni] = (v8f){0.f, 0.f, 0.f, 0.f, 0.f, 0.f, 0.f, 0.f};

    auto compute = [&](int buf) {
        const uint32_t* Ahi = &lds[buf][0][0];
        const uint32_t* Alo = &lds[buf][1][0];
        const uint32_t* Bhi = &lds[buf][2][0];
        const uint32_t* Blo = &lds[buf][3][0];
        v16bf ah[2], al[2], bh[4], bl[4];
        #pragma unroll
        for (int mi = 0; mi < 2; ++mi) {
            ah[mi] = load_frag(Ahi, wm * 32 + mi * 16, lane);
            al[mi] = load_frag(Alo, wm * 32 + mi * 16, lane);
        }
        #pragma unroll
        for (int ni = 0; ni < 4; ++ni) {
            bh[ni] = load_frag(Bhi, wn * 64 + ni * 16, lane);
            bl[ni] = load_frag(Blo, wn * 64 + ni * 16, lane);
        }
        #pragma unroll
        for (int mi = 0; mi < 2; ++mi) {
            #pragma unroll
            for (int ni = 0; ni < 4; ++ni) {
                acc[mi][ni] = __builtin_amdgcn_wmma_f32_16x16x32_bf16(
                    false, ah[mi], false, bh[ni], (short)0, acc[mi][ni], false, false);
                acc[mi][ni] = __builtin_amdgcn_wmma_f32_16x16x32_bf16(
                    false, ah[mi], false, bl[ni], (short)0, acc[mi][ni], false, false);
                acc[mi][ni] = __builtin_amdgcn_wmma_f32_16x16x32_bf16(
                    false, al[mi], false, bh[ni], (short)0, acc[mi][ni], false, false);
            }
        }
    };

    // ---- prologue: stage chunk 0 into buffer 0 ----
    loadA(0);
    if constexpr (USE_WS) stageB_async(0, 0); else loadB(0);
    storeA(0);
    if constexpr (!USE_WS) storeB(0);
    if constexpr (USE_WS) wait_async0();
    __syncthreads();

    // ---- main loop: double buffered, one barrier per chunk ----
    for (int c = 0; c < NCHUNK; ++c) {
        const int buf = c & 1;
        const bool has_next = (c + 1) < NCHUNK;
        const int kn = (c + 1) * BK;
        if (has_next) {
            loadA(kn);                                   // global loads issue early
            if constexpr (USE_WS) stageB_async(buf ^ 1, kn);
            else loadB(kn);
        }
        compute(buf);                                    // 24 WMMAs hide the latency
        if (has_next) {
            storeA(buf ^ 1);                             // convert + ds_store after WMMA
            if constexpr (!USE_WS) storeB(buf ^ 1);
            if constexpr (USE_WS) wait_async0();         // my async writes visible
        }
        __syncthreads();
    }

    // ---- epilogue: add bias, store fp32 ----
    // C/D layout: lane = N (0..15 twice); VGPR r -> M = r + (lane>=16 ? 8 : 0)
    const int colHalf = lane & 15;
    const int mOff    = (lane >> 4) << 3;
    #pragma unroll
    for (int ni = 0; ni < 4; ++ni) {
        const int col = rowN0 + wn * 64 + ni * 16 + colHalf;
        const float bv = bias[col];
        #pragma unroll
        for (int mi = 0; mi < 2; ++mi) {
            const int row0 = rowM0 + wm * 32 + mi * 16 + mOff;
            float* op = out + (size_t)row0 * D_DIM + col;
            #pragma unroll
            for (int r = 0; r < 8; ++r) {
                op[(size_t)r * D_DIM] = acc[mi][ni][r] + bv;
            }
        }
    }
}

extern "C" void kernel_launch(void* const* d_in, const int* in_sizes, int n_in,
                              void* d_out, int out_size, void* d_ws, size_t ws_size,
                              hipStream_t stream) {
    // input order: x, cond, mask, W_shared, W_expert, W_gate, bias
    const float* x        = (const float*)d_in[0];
    const float* W_shared = (const float*)d_in[3];
    const float* W_expert = (const float*)d_in[4];
    const float* bias     = (const float*)d_in[6];
    float* out = (float*)d_out;

    dim3 grid(D_DIM / BN, T_DIM / BM);               // 8 x 128 workgroups
    dim3 block(256);                                 // 8 waves (wave32)

    const size_t need = (size_t)D_DIM * D_DIM * 2 * sizeof(uint16_t);  // 4 MB hi+lo
    if (ws_size >= need) {
        uint32_t* wh = (uint32_t*)d_ws;
        uint32_t* wl = wh + (D_DIM * (size_t)D_DIM / 2);
        prep_w_kernel<<<(D_DIM * D_DIM / 2) / 256, 256, 0, stream>>>(W_shared, W_expert, wh, wl);
        moe_gemm_kernel<true><<<grid, block, 0, stream>>>(x, nullptr, nullptr, wh, wl, bias, out);
    } else {
        moe_gemm_kernel<false><<<grid, block, 0, stream>>>(x, W_shared, W_expert,
                                                           nullptr, nullptr, bias, out);
    }
}
